// DeformableTransformer_60962765800058
// MI455X (gfx1250) — matrix operand order repaired and checked by reading, not compiled, not run on hardware
//
#include <hip/hip_runtime.h>

// ---------------- problem constants ----------------
#define N_    16
#define LQ_   300
#define D_    256
#define H_    8
#define DH_   32
#define L_    3
#define P_    4
#define DFF_  1024
#define NL_   6
#define T_    7168
#define NQ_   (N_*LQ_)      // 4800
#define NT_   (N_*T_)       // 114688
#define NH_   (N_*H_)       // 128
#define LQR_  304           // LQ padded to tile multiple (rows of S/P)
#define LQP_  320           // LQ padded to multiple of 32 (cols of S/P, K of P@V)
#define TWO_PI 6.283185307179586f

typedef __bf16 bf16;
typedef __attribute__((ext_vector_type(16))) __bf16 v16bf;
typedef __attribute__((ext_vector_type(8)))  __bf16 v8bf;
typedef __attribute__((ext_vector_type(8)))  float  v8f;

static __device__ __forceinline__ v8f wmma_bf16(v16bf a, v16bf b, v8f c) {
  return __builtin_amdgcn_wmma_f32_16x16x32_bf16(false, a, false, b, (short)0, c, false, false);
}

// A-matrix fragment (16x32, 16-bit): lane holds row lane%16.
// lanes 0-15: elems = K[0..15]; lanes 16-31: elems = K[16..31]
static __device__ __forceinline__ v16bf ld_a_frag(const bf16* __restrict__ p, int lhi) {
  const int base = lhi ? 16 : 0;
  v8bf lo = *(const v8bf*)(p + base);
  v8bf hi = *(const v8bf*)(p + base + 8);
  v16bf r;
#pragma unroll
  for (int i = 0; i < 8; ++i) { r[i] = lo[i]; r[i + 8] = hi[i]; }
  return r;
}

// B-matrix fragment (32x16, 16-bit): lane holds column lane%16.
// lanes 0-15: elems = K[0..15]; lanes 16-31: elems = K[16..31]  (contiguous 32B)
static __device__ __forceinline__ v16bf ld_b_frag(const bf16* __restrict__ p, int lhi) {
  return *(const v16bf*)(p + (lhi ? 16 : 0));
}

// ---------------- elementwise helpers ----------------
__global__ void convert_f32_bf16(const float* __restrict__ s, bf16* __restrict__ d, int n) {
  int i = blockIdx.x * blockDim.x + threadIdx.x;
  if (i < n) d[i] = (bf16)s[i];
}
__global__ void mul_f32(const float* __restrict__ a, const float* __restrict__ b,
                        float* __restrict__ o, int n) {
  int i = blockIdx.x * blockDim.x + threadIdx.x;
  if (i < n) o[i] = a[i] * b[i];
}
__global__ void add_convert_bf16(const float* __restrict__ a, const float* __restrict__ b,
                                 bf16* __restrict__ o, int n) {
  int i = blockIdx.x * blockDim.x + threadIdx.x;
  if (i < n) o[i] = (bf16)(a[i] + b[i]);
}

// sine positional embedding -> (NQ, 2D) bf16.  ref = (c, exp(w))
__global__ void sine_embed_kernel(const float* __restrict__ seg, bf16* __restrict__ out) {
  int i = blockIdx.x * blockDim.x + threadIdx.x;
  if (i >= NQ_ * 2 * D_) return;
  int nq = i >> 9;            // /512
  int k  = i & 511;
  float c = seg[nq * 2 + 0];
  float w = __expf(seg[nq * 2 + 1]);
  int j = k & 255;
  float base = (k < 256) ? c : w;
  float dimt = __powf(10000.f, (float)(2 * (j >> 1)) * (1.f / 256.f));
  float v = base * TWO_PI / dimt;
  float r = ((j & 1) == 0) ? __sinf(v) : __cosf(v);
  out[i] = (bf16)r;
}

// ---------------- main WMMA GEMM: C(MxNc) = A(MxK) @ W(NcxK)^T + bias ----------------
// block = 8 waves, tile 128 rows x (NF*16) cols; wave tile 16 x (NF*16).
// NF and the output mode are compile-time so both the K loop and the epilogue
// are branch-free (accumulators stay pinned, single store per element).
template <int NF, bool OUT_BF16>
__global__ void __launch_bounds__(256)
gemm_bf16_kernel(const bf16* __restrict__ A, int lda,
                 const bf16* __restrict__ W, int ldw,
                 const float* __restrict__ bias,
                 float* __restrict__ Cf, bf16* __restrict__ Cb, int ldc,
                 int M, int K, int relu,
                 const unsigned char* __restrict__ rowmask)
{
  const int wave = threadIdx.x >> 5;
  const int lane = threadIdx.x & 31;
  const int row0 = blockIdx.x * 128 + wave * 16;
  const int col0 = blockIdx.y * (NF * 16);
  if (row0 >= M) return;                       // wave-uniform (M % 16 == 0 always here)
  const int lrow = lane & 15;
  const int lhi  = lane >> 4;
  v8f zero = {0.f,0.f,0.f,0.f,0.f,0.f,0.f,0.f};
  v8f acc[NF];
#pragma unroll
  for (int f = 0; f < NF; ++f) acc[f] = zero;
  const bf16* Arow = A + (size_t)(row0 + lrow) * lda;
  const bf16* Wrow0 = W + (size_t)(col0 + lrow) * ldw;
  for (int k = 0; k < K; k += 32) {
    v16bf a = ld_a_frag(Arow + k, lhi);
#pragma unroll
    for (int f = 0; f < NF; ++f) {
      v16bf b = ld_b_frag(Wrow0 + (size_t)(f * 16) * ldw + k, lhi);
      acc[f] = wmma_bf16(a, b, acc[f]);
    }
  }
  float bv[NF];
#pragma unroll
  for (int f = 0; f < NF; ++f) bv[f] = bias ? bias[col0 + f * 16 + lrow] : 0.f;
#pragma unroll
  for (int v = 0; v < 8; ++v) {
    const int r = row0 + v + lhi * 8;          // D: vgpr v = row (m%8), half by lane>=16
    const float msk = rowmask ? (rowmask[r] ? 0.f : 1.f) : 1.f;
#pragma unroll
    for (int f = 0; f < NF; ++f) {
      const int c = col0 + f * 16 + lrow;      // D: lane%16 = output column
      float x = (acc[f][v] + bv[f]) * msk;
      if (relu) x = fmaxf(x, 0.f);
      const size_t idx = (size_t)r * ldc + c;
      if (OUT_BF16) Cb[idx] = (bf16)x; else Cf[idx] = x;
    }
  }
}

// ---------------- self-attention: S = Q K^T / sqrt(Dh)  (WMMA, Dh=32 -> one MAC) ----
// qk: (NQ, 512) bf16 (cols 0..255 = q heads, 256..511 = k heads).
// S: (NH, LQR, LQP) f32, rows/cols 300..303 are padding -> unconditional stores.
__global__ void __launch_bounds__(256)
attn_scores_kernel(const bf16* __restrict__ qk, float* __restrict__ S)
{
  const int wave = threadIdx.x >> 5, lane = threadIdx.x & 31;
  const int tj = blockIdx.y * 8 + wave;
  if (tj >= 19) return;
  const int ti = blockIdx.x;
  const int nh = blockIdx.z, n = nh >> 3, h = nh & 7;
  const int lrow = lane & 15, lhi = lane >> 4;
  const int i = min(ti * 16 + lrow, LQ_ - 1);
  const int j = min(tj * 16 + lrow, LQ_ - 1);
  const bf16* qrow = qk + ((size_t)(n * LQ_ + i)) * 512 + h * DH_;
  const bf16* krow = qk + ((size_t)(n * LQ_ + j)) * 512 + 256 + h * DH_;
  v16bf a = ld_a_frag(qrow, lhi);
  v16bf b = ld_b_frag(krow, lhi);
  v8f c = {0.f,0.f,0.f,0.f,0.f,0.f,0.f,0.f};
  c = wmma_bf16(a, b, c);
  float* srow = S + ((size_t)nh * LQR_ + ti * 16 + lhi * 8) * LQP_ + tj * 16 + lrow;
#pragma unroll
  for (int v = 0; v < 8; ++v)
    srow[(size_t)v * LQP_] = c[v] * 0.17677669529663687f;   // 1/sqrt(32)
}

// row softmax over j<LQ, write bf16 probs padded with zeros to LQP.
// Runs over all LQR rows (padding rows hold finite garbage, never consumed).
__global__ void __launch_bounds__(256)
attn_softmax_kernel(const float* __restrict__ S, bf16* __restrict__ Pb)
{
  const int row = blockIdx.x * 8 + (threadIdx.x >> 5);
  const int lane = threadIdx.x & 31;
  if (row >= NH_ * LQR_) return;
  const float* s = S + (size_t)row * LQP_;
  float mx = -1e30f;
  for (int j = lane; j < LQ_; j += 32) mx = fmaxf(mx, s[j]);
  for (int o = 16; o > 0; o >>= 1) mx = fmaxf(mx, __shfl_xor(mx, o, 32));
  float sum = 0.f;
  for (int j = lane; j < LQ_; j += 32) sum += __expf(s[j] - mx);
  for (int o = 16; o > 0; o >>= 1) sum += __shfl_xor(sum, o, 32);
  const float inv = 1.f / sum;
  bf16* p = Pb + (size_t)row * LQP_;
  for (int j = lane; j < LQP_; j += 32)
    p[j] = (bf16)((j < LQ_) ? __expf(s[j] - mx) * inv : 0.f);
}

// O = P @ V  (WMMA over K=LQP in chunks of 32).  V: (NQ, 256) bf16, O: (NQ,256) bf16
__global__ void __launch_bounds__(256)
attn_pv_kernel(const bf16* __restrict__ Pm, const bf16* __restrict__ Vb,
               bf16* __restrict__ Ob)
{
  const int wave = threadIdx.x >> 5, lane = threadIdx.x & 31;
  const int ti = blockIdx.x * 8 + wave;
  if (ti >= 19) return;
  const int td = blockIdx.y;                     // 0..1 (two 16-col halves of Dh=32)
  const int nh = blockIdx.z, n = nh >> 3, h = nh & 7;
  const int lrow = lane & 15, lhi = lane >> 4;
  const int i = min(ti * 16 + lrow, LQ_ - 1);
  const bf16* prow = Pm + ((size_t)nh * LQR_ + i) * LQP_;
  const int dcol = h * DH_ + td * 16 + lrow;
  const bf16* vcol = Vb + (size_t)n * LQ_ * D_ + dcol;
  v8f acc = {0.f,0.f,0.f,0.f,0.f,0.f,0.f,0.f};
  // first 9 chunks: V row index always < 300, no clamp needed
  for (int k = 0; k < LQP_ - 32; k += 32) {
    v16bf a = ld_a_frag(prow + k, lhi);
    const int kb = k + (lhi ? 16 : 0);
    v16bf b;
#pragma unroll
    for (int e = 0; e < 16; ++e) b[e] = vcol[(size_t)(kb + e) * D_];
    acc = wmma_bf16(a, b, acc);
  }
  {  // last chunk: clamp V row (P is zero beyond LQ, so value is irrelevant)
    const int k = LQP_ - 32;
    v16bf a = ld_a_frag(prow + k, lhi);
    const int kb = k + (lhi ? 16 : 0);
    v16bf b;
#pragma unroll
    for (int e = 0; e < 16; ++e) b[e] = vcol[(size_t)min(kb + e, LQ_ - 1) * D_];
    acc = wmma_bf16(a, b, acc);
  }
#pragma unroll
  for (int v = 0; v < 8; ++v) {
    const int m = ti * 16 + v + lhi * 8;
    if (m < LQ_) Ob[((size_t)(n * LQ_ + m)) * D_ + dcol] = (bf16)acc[v];
  }
}

// ---------------- deformable attention pieces ----------------
// softmax over the 12 (L*P) attention-weight logits for each (n,q,h), in place
__global__ void aw_softmax_kernel(float* __restrict__ aw)
{
  int t = blockIdx.x * blockDim.x + threadIdx.x;
  if (t >= NQ_ * H_) return;
  int nq = t >> 3, h = t & 7;
  float* a = aw + (size_t)nq * (H_ * L_ * P_) + h * (L_ * P_);
  float mx = -1e30f;
  for (int i = 0; i < L_ * P_; ++i) mx = fmaxf(mx, a[i]);
  float s = 0.f;
  for (int i = 0; i < L_ * P_; ++i) { float e = __expf(a[i] - mx); a[i] = e; s += e; }
  float inv = 1.f / s;
  for (int i = 0; i < L_ * P_; ++i) a[i] *= inv;
}

// bilinear sampling + weighted sum. block = (n,q); wave = head; lane = channel d.
__global__ void __launch_bounds__(256)
deform_sample_kernel(const float* __restrict__ seg, const float* __restrict__ dur,
                     const float* __restrict__ ratios,
                     const float* __restrict__ offs, const float* __restrict__ aws,
                     const bf16* __restrict__ val,
                     const int* __restrict__ tlens, const int* __restrict__ lstart,
                     bf16* __restrict__ out)
{
  const int nq = blockIdx.x;
  const int h = threadIdx.x >> 5;
  const int d = threadIdx.x & 31;
  const int n = nq / LQ_;
  const float c = seg[nq * 2 + 0];
  const float wdt = __expf(seg[nq * 2 + 1]);
  const float du = dur[n];
  float acc = 0.f;
  for (int l = 0; l < L_; ++l) {
    const float vr = ratios[n * L_ + l];
    const float r0 = c / du * vr;
    const float r1 = wdt / du * vr;
    const int tl = tlens[l];
    const int st = lstart[l];
#pragma unroll
    for (int p = 0; p < P_; ++p) {
      const int idx = h * (L_ * P_) + l * P_ + p;
      const float off = offs[(size_t)nq * (H_ * L_ * P_) + idx];
      const float w   = aws [(size_t)nq * (H_ * L_ * P_) + idx];
      const float loc = r0 + off * (1.f / P_) * r1 * 0.5f;
      const float x = loc * (float)tl - 0.5f;
      const float x0f = floorf(x);
      const float fr = x - x0f;
      const int x0 = (int)x0f;
      const bool m0 = (x0 >= 0) && (x0 < tl);
      const bool m1 = (x0 + 1 >= 0) && (x0 + 1 < tl);
      const int i0 = st + min(max(x0, 0), tl - 1);
      const int i1 = st + min(max(x0 + 1, 0), tl - 1);
      const float g0 = m0 ? (float)val[((size_t)n * T_ + i0) * D_ + h * DH_ + d] : 0.f;
      const float g1 = m1 ? (float)val[((size_t)n * T_ + i1) * D_ + h * DH_ + d] : 0.f;
      acc += w * (g0 * (1.f - fr) + g1 * fr);
    }
  }
  out[(size_t)nq * D_ + h * DH_ + d] = (bf16)acc;
}

// ---------------- residual + LayerNorm (row = block of 256 threads) -------------
__global__ void __launch_bounds__(256)
add_ln_kernel(const float* __restrict__ a, const float* __restrict__ b,
              const float* __restrict__ g, const float* __restrict__ be,
              float* __restrict__ out, bf16* __restrict__ outbf)
{
  __shared__ float sm[8];
  const int row = blockIdx.x, t = threadIdx.x;
  const size_t idx = (size_t)row * D_ + t;
  const float v = a[idx] + b[idx];
  float s = v;
  for (int o = 16; o > 0; o >>= 1) s += __shfl_xor(s, o, 32);
  if ((t & 31) == 0) sm[t >> 5] = s;
  __syncthreads();
  float tot = 0.f;
#pragma unroll
  for (int i = 0; i < 8; ++i) tot += sm[i];
  const float mean = tot * (1.f / D_);
  const float dv = v - mean;
  __syncthreads();
  float s2 = dv * dv;
  for (int o = 16; o > 0; o >>= 1) s2 += __shfl_xor(s2, o, 32);
  if ((t & 31) == 0) sm[t >> 5] = s2;
  __syncthreads();
  float tv = 0.f;
#pragma unroll
  for (int i = 0; i < 8; ++i) tv += sm[i];
  const float var = tv * (1.f / D_);
  const float y = dv * rsqrtf(var + 1e-5f) * g[t] + be[t];
  out[idx] = y;
  if (outbf) outbf[idx] = (bf16)y;
}

// ---------------- host orchestration ----------------
static inline int cdiv(int a, int b) { return (a + b - 1) / b; }

extern "C" void kernel_launch(void* const* d_in, const int* in_sizes, int n_in,
                              void* d_out, int out_size, void* d_ws, size_t ws_size,
                              hipStream_t stream) {
  const float* tgt      = (const float*)d_in[0];
  const float* seg      = (const float*)d_in[1];
  const float* dur      = (const float*)d_in[2];
  const float* src      = (const float*)d_in[3];
  const int*   tlens    = (const int*)d_in[4];
  const int*   lstart   = (const int*)d_in[5];
  const float* ratios   = (const float*)d_in[6];
  const unsigned char* padmask = (const unsigned char*)d_in[7];
  const float* grid_w0  = (const float*)d_in[8];
  const float* grid_b0  = (const float*)d_in[9];
  const float* grid_w1  = (const float*)d_in[10];
  const float* grid_b1  = (const float*)d_in[11];
  const float* qs_w0    = (const float*)d_in[12];
  const float* qs_b0    = (const float*)d_in[13];
  const float* qs_w1    = (const float*)d_in[14];
  const float* qs_b1    = (const float*)d_in[15];
  const float* sa_in_w  = (const float*)d_in[16];
  const float* sa_in_b  = (const float*)d_in[17];
  const float* sa_out_w = (const float*)d_in[18];
  const float* sa_out_b = (const float*)d_in[19];
  const float* n1_g = (const float*)d_in[20];
  const float* n1_b = (const float*)d_in[21];
  const float* n2_g = (const float*)d_in[22];
  const float* n2_b = (const float*)d_in[23];
  const float* n3_g = (const float*)d_in[24];
  const float* n3_b = (const float*)d_in[25];
  const float* off_w  = (const float*)d_in[26];
  const float* off_b  = (const float*)d_in[27];
  const float* aw_w   = (const float*)d_in[28];
  const float* aw_b   = (const float*)d_in[29];
  const float* val_w  = (const float*)d_in[30];
  const float* val_b  = (const float*)d_in[31];
  const float* outp_w = (const float*)d_in[32];
  const float* outp_b = (const float*)d_in[33];
  const float* ffn_w1 = (const float*)d_in[34];
  const float* ffn_b1 = (const float*)d_in[35];
  const float* ffn_w2 = (const float*)d_in[36];
  const float* ffn_b2 = (const float*)d_in[37];

  // workspace bump allocator (256B aligned)
  char* base = (char*)d_ws;
  size_t off = 0;
  auto alloc = [&](size_t bytes) -> void* {
    void* p = base + off;
    off = (off + bytes + 255) & ~(size_t)255;
    return p;
  };
  auto abf = [&](size_t n) -> bf16*  { return (bf16*)alloc(n * sizeof(bf16)); };
  auto af  = [&](size_t n) -> float* { return (float*)alloc(n * sizeof(float)); };

  // bf16 weights
  bf16* grid_w0_bf = abf((size_t)D_ * 2 * D_);
  bf16* grid_w1_bf = abf((size_t)D_ * D_);
  bf16* qs_w0_bf   = abf((size_t)D_ * D_);
  bf16* qs_w1_bf   = abf((size_t)D_ * D_);
  bf16* sa_in_w_bf = abf((size_t)NL_ * 3 * D_ * D_);
  bf16* sa_out_w_bf= abf((size_t)NL_ * D_ * D_);
  bf16* off_w_bf   = abf((size_t)NL_ * H_ * L_ * P_ * D_);
  bf16* aw_w_bf    = abf((size_t)NL_ * H_ * L_ * P_ * D_);
  bf16* val_w_bf   = abf((size_t)NL_ * D_ * D_);
  bf16* outp_w_bf  = abf((size_t)NL_ * D_ * D_);
  bf16* ffn_w1_bf  = abf((size_t)NL_ * DFF_ * D_);
  bf16* ffn_w2_bf  = abf((size_t)NL_ * D_ * DFF_);
  // activations
  bf16* src_bf   = abf((size_t)NT_ * D_);
  bf16* sine_bf  = abf((size_t)NQ_ * 2 * D_);
  bf16* gridh_bf = abf((size_t)NQ_ * D_);
  float* raw_qpos= af((size_t)NQ_ * D_);
  float* out_f   = af((size_t)NQ_ * D_);
  bf16* out_bf   = abf((size_t)NQ_ * D_);
  bf16* tmp_bf   = abf((size_t)NQ_ * D_);
  float* qs_out  = af((size_t)NQ_ * D_);
  float* qpos_f  = af((size_t)NQ_ * D_);
  bf16* q_bf     = abf((size_t)NQ_ * D_);
  bf16* qk_bf    = abf((size_t)NQ_ * 2 * D_);
  bf16* v_bf     = abf((size_t)NQ_ * D_);
  float* S_f     = af((size_t)NH_ * LQR_ * LQP_);
  bf16* P_bf     = abf((size_t)NH_ * LQR_ * LQP_);
  bf16* attn_bf  = abf((size_t)NQ_ * D_);
  float* mha_f   = af((size_t)NQ_ * D_);
  float* off_f   = af((size_t)NQ_ * H_ * L_ * P_);
  float* aw_f    = af((size_t)NQ_ * H_ * L_ * P_);
  bf16* val_bf   = abf((size_t)NT_ * D_);
  bf16* samp_bf  = abf((size_t)NQ_ * D_);
  float* ca_f    = af((size_t)NQ_ * D_);
  bf16* ffh_bf   = abf((size_t)NQ_ * DFF_);
  float* ffo_f   = af((size_t)NQ_ * D_);

  auto cvt = [&](const float* s, bf16* d, size_t n) {
    convert_f32_bf16<<<cdiv((int)n, 256), 256, 0, stream>>>(s, d, (int)n);
  };
  auto gemm = [&](const bf16* A, int lda, const bf16* W, int ldw, const float* bias,
                  float* Cf, bf16* Cb, int ldc, int M, int Nc, int K, int relu,
                  const unsigned char* rowmask) {
    if (Nc % 64 == 0) {
      dim3 g(cdiv(M, 128), Nc / 64);
      if (Cb) gemm_bf16_kernel<4, true ><<<g, 256, 0, stream>>>(A, lda, W, ldw, bias,
                                  Cf, Cb, ldc, M, K, relu, rowmask);
      else    gemm_bf16_kernel<4, false><<<g, 256, 0, stream>>>(A, lda, W, ldw, bias,
                                  Cf, Cb, ldc, M, K, relu, rowmask);
    } else {
      dim3 g(cdiv(M, 128), Nc / 32);
      if (Cb) gemm_bf16_kernel<2, true ><<<g, 256, 0, stream>>>(A, lda, W, ldw, bias,
                                  Cf, Cb, ldc, M, K, relu, rowmask);
      else    gemm_bf16_kernel<2, false><<<g, 256, 0, stream>>>(A, lda, W, ldw, bias,
                                  Cf, Cb, ldc, M, K, relu, rowmask);
    }
  };

  // ---- one-time conversions ----
  cvt(grid_w0, grid_w0_bf, (size_t)D_ * 2 * D_);
  cvt(grid_w1, grid_w1_bf, (size_t)D_ * D_);
  cvt(qs_w0, qs_w0_bf, (size_t)D_ * D_);
  cvt(qs_w1, qs_w1_bf, (size_t)D_ * D_);
  cvt(sa_in_w, sa_in_w_bf, (size_t)NL_ * 3 * D_ * D_);
  cvt(sa_out_w, sa_out_w_bf, (size_t)NL_ * D_ * D_);
  cvt(off_w, off_w_bf, (size_t)NL_ * H_ * L_ * P_ * D_);
  cvt(aw_w, aw_w_bf, (size_t)NL_ * H_ * L_ * P_ * D_);
  cvt(val_w, val_w_bf, (size_t)NL_ * D_ * D_);
  cvt(outp_w, outp_w_bf, (size_t)NL_ * D_ * D_);
  cvt(ffn_w1, ffn_w1_bf, (size_t)NL_ * DFF_ * D_);
  cvt(ffn_w2, ffn_w2_bf, (size_t)NL_ * D_ * DFF_);
  cvt(src, src_bf, (size_t)NT_ * D_);

  // out = tgt (f32 + bf16 copies)
  hipMemcpyAsync(out_f, tgt, (size_t)NQ_ * D_ * sizeof(float),
                 hipMemcpyDeviceToDevice, stream);
  cvt(out_f, out_bf, (size_t)NQ_ * D_);

  // loop-invariant: sine embed -> grid MLP -> raw_qpos
  sine_embed_kernel<<<cdiv(NQ_ * 2 * D_, 256), 256, 0, stream>>>(seg, sine_bf);
  gemm(sine_bf, 2 * D_, grid_w0_bf, 2 * D_, grid_b0, nullptr, gridh_bf, D_,
       NQ_, D_, 2 * D_, 1, nullptr);
  gemm(gridh_bf, D_, grid_w1_bf, D_, grid_b1, raw_qpos, nullptr, D_,
       NQ_, D_, D_, 0, nullptr);

  const int HLP = H_ * L_ * P_;
  for (int l = 0; l < NL_; ++l) {
    // ---- query pos MLP: qpos = MLP(out) * raw_qpos ----
    gemm(out_bf, D_, qs_w0_bf, D_, qs_b0, nullptr, tmp_bf, D_, NQ_, D_, D_, 1, nullptr);
    gemm(tmp_bf, D_, qs_w1_bf, D_, qs_b1, qs_out, nullptr, D_, NQ_, D_, D_, 0, nullptr);
    mul_f32<<<cdiv(NQ_ * D_, 256), 256, 0, stream>>>(qs_out, raw_qpos, qpos_f, NQ_ * D_);
    add_convert_bf16<<<cdiv(NQ_ * D_, 256), 256, 0, stream>>>(out_f, qpos_f, q_bf, NQ_ * D_);

    // ---- self attention ----
    gemm(q_bf, D_, sa_in_w_bf + (size_t)l * 3 * D_ * D_, D_,
         sa_in_b + (size_t)l * 3 * D_, nullptr, qk_bf, 2 * D_,
         NQ_, 2 * D_, D_, 0, nullptr);                             // q,k projections
    gemm(out_bf, D_, sa_in_w_bf + (size_t)l * 3 * D_ * D_ + (size_t)2 * D_ * D_, D_,
         sa_in_b + (size_t)l * 3 * D_ + 2 * D_, nullptr, v_bf, D_,
         NQ_, D_, D_, 0, nullptr);                                 // v projection
    attn_scores_kernel<<<dim3(19, 3, NH_), 256, 0, stream>>>(qk_bf, S_f);
    attn_softmax_kernel<<<cdiv(NH_ * LQR_, 8), 256, 0, stream>>>(S_f, P_bf);
    attn_pv_kernel<<<dim3(3, 2, NH_), 256, 0, stream>>>(P_bf, v_bf, attn_bf);
    gemm(attn_bf, D_, sa_out_w_bf + (size_t)l * D_ * D_, D_,
         sa_out_b + (size_t)l * D_, mha_f, nullptr, D_, NQ_, D_, D_, 0, nullptr);
    add_ln_kernel<<<NQ_, 256, 0, stream>>>(out_f, mha_f,
        n1_g + (size_t)l * D_, n1_b + (size_t)l * D_, out_f, (bf16*)nullptr);

    // ---- deformable cross attention ----
    add_convert_bf16<<<cdiv(NQ_ * D_, 256), 256, 0, stream>>>(out_f, qpos_f, q_bf, NQ_ * D_);
    gemm(q_bf, D_, off_w_bf + (size_t)l * HLP * D_, D_,
         off_b + (size_t)l * HLP, off_f, nullptr, HLP, NQ_, HLP, D_, 0, nullptr);
    gemm(q_bf, D_, aw_w_bf + (size_t)l * HLP * D_, D_,
         aw_b + (size_t)l * HLP, aw_f, nullptr, HLP, NQ_, HLP, D_, 0, nullptr);
    aw_softmax_kernel<<<cdiv(NQ_ * H_, 256), 256, 0, stream>>>(aw_f);
    gemm(src_bf, D_, val_w_bf + (size_t)l * D_ * D_, D_,
         val_b + (size_t)l * D_, nullptr, val_bf, D_, NT_, D_, D_, 0, padmask);
    deform_sample_kernel<<<NQ_, 256, 0, stream>>>(seg, dur, ratios, off_f, aw_f,
                                                  val_bf, tlens, lstart, samp_bf);
    gemm(samp_bf, D_, outp_w_bf + (size_t)l * D_ * D_, D_,
         outp_b + (size_t)l * D_, ca_f, nullptr, D_, NQ_, D_, D_, 0, nullptr);
    add_ln_kernel<<<NQ_, 256, 0, stream>>>(out_f, ca_f,
        n2_g + (size_t)l * D_, n2_b + (size_t)l * D_, out_f, out_bf);

    // ---- FFN ----
    gemm(out_bf, D_, ffn_w1_bf + (size_t)l * DFF_ * D_, D_,
         ffn_b1 + (size_t)l * DFF_, nullptr, ffh_bf, DFF_, NQ_, DFF_, D_, 1, nullptr);
    gemm(ffh_bf, DFF_, ffn_w2_bf + (size_t)l * D_ * DFF_, DFF_,
         ffn_b2 + (size_t)l * D_, ffo_f, nullptr, D_, NQ_, D_, DFF_, 0, nullptr);
    add_ln_kernel<<<NQ_, 256, 0, stream>>>(out_f, ffo_f,
        n3_g + (size_t)l * D_, n3_b + (size_t)l * D_, out_f, out_bf);
  }

  hipMemcpyAsync(d_out, out_f, (size_t)NQ_ * D_ * sizeof(float),
                 hipMemcpyDeviceToDevice, stream);
}
